// TensorProductRescale_9818295239480
// MI455X (gfx1250) — matrix operand order repaired and checked by reading, not compiled, not run on hardware
//
#include <hip/hip_runtime.h>

typedef __attribute__((ext_vector_type(4)))  __bf16 v4bf;
typedef __attribute__((ext_vector_type(8)))  __bf16 v8bf;
typedef __attribute__((ext_vector_type(16))) __bf16 v16bf;
typedef __attribute__((ext_vector_type(8)))  float  v8f;

constexpr int   ROW    = 320;                  // floats per x / out row (128 + 3*64)
constexpr int   MUL0   = 128;
constexpr float C_IS3  = 0.57735026918962576f; // 1/sqrt(3)
constexpr float C_IS2  = 0.70710678118654752f; // 1/sqrt(2)

constexpr int FS    = 712;  // per-sample bf16 feature stride (704 used + 8 pad)
constexpr int WA_S  = 200;  // K-stride of WAt  (192 + 8 pad)
constexpr int WB_S  = 136;  // K-stride of WBt/WCt (128 + 8 pad)
constexpr int WAVES = 8;

__device__ __forceinline__ v16bf cat8(v8bf a, v8bf b) {
  return __builtin_shufflevector(a, b, 0,1,2,3,4,5,6,7,8,9,10,11,12,13,14,15);
}
__device__ __forceinline__ v8f wmma_bf16(v16bf a, v16bf b, v8f c) {
  return __builtin_amdgcn_wmma_f32_16x16x32_bf16(false, a, false, b, (short)0, c, false, false);
}

__global__ __launch_bounds__(256, 1)
void tpr_kernel(const float* __restrict__ x,    const float* __restrict__ y,
                const float* __restrict__ w000, const float* __restrict__ w011,
                const float* __restrict__ w101, const float* __restrict__ w110,
                const float* __restrict__ w111, const float* __restrict__ bias,
                float* __restrict__ out, int N, int ntiles, int iters)
{
  extern __shared__ char smemRaw[];
  __bf16* WAt   = (__bf16*)smemRaw;                 // [col<128][K<192]  K-major
  __bf16* WBt   = WAt + 128 * WA_S;                 // [col<64 ][K<128]
  __bf16* WCt   = WBt + 64 * WB_S;                  // [col<64 ][K<128]
  float*  sBias = (float*)(WCt + 64 * WB_S);        // 128
  float*  sY    = sBias + 128;                      // WAVES*16*4
  __bf16* sFeat = (__bf16*)(sY + WAVES * 64);       // WAVES*16*FS

  const int tid = threadIdx.x;

  // ---- Phase 0: weights -> LDS bf16, transposed to K-major, scales folded ----
  for (int i = tid; i < 128 * 192; i += 256) {
    int col = i / 192, K = i - col * 192;
    float v = (K < 128) ? w000[K * 128 + col] : C_IS3 * w110[(K - 128) * 128 + col];
    WAt[col * WA_S + K] = (__bf16)v;
  }
  for (int i = tid; i < 64 * 128; i += 256) {
    int col = i / 128, K = i - col * 128;
    WBt[col * WB_S + K] = (__bf16)w011[K * 64 + col];
    float v = (K < 64) ? w101[K * 64 + col] : C_IS2 * w111[(K - 64) * 64 + col];
    WCt[col * WB_S + K] = (__bf16)v;
  }
  if (tid < 128) sBias[tid] = bias[tid];
  __syncthreads();

  const int wave = tid >> 5, lane = tid & 31;
  const int m16 = lane & 15, hi = lane >> 4;
  __bf16* feat = sFeat + wave * 16 * FS;
  float*  yW   = sY + wave * 64;
  const int totalWaves = gridDim.x * WAVES;

  for (int it = 0; it < iters; ++it) {
    // Wave-uniform by construction; force into SGPR so control flow scalarizes.
    const int  tile   = __builtin_amdgcn_readfirstlane(
                            blockIdx.x * WAVES + wave + it * totalWaves);
    const bool active = tile < ntiles;

    // ---- Phase 1: build bf16 A-features for 16 samples (2 lanes per sample) ----
    if (active) {
      const int n0 = tile * 16;
      const int m = lane >> 1, role = lane & 1;
      int n = n0 + m; if (n >= N) n = N - 1;
      const float4 yv = ((const float4*)y)[n];
      const float y0 = yv.x, b0 = yv.y, b1 = yv.z, b2 = yv.w;
      __bf16* fr = feat + m * FS;
      if (role == 0) {
        ((float4*)yW)[m] = yv;
        const float4* xr = (const float4*)(x + (size_t)n * ROW);
        #pragma unroll 4
        for (int i = 0; i < 32; ++i) {
          float4 f = xr[i];
          v4bf a = { (__bf16)(f.x * y0), (__bf16)(f.y * y0),
                     (__bf16)(f.z * y0), (__bf16)(f.w * y0) };
          *(v4bf*)(fr + 4 * i) = a;                 // [0,128): x0*y0
          v4bf b = { (__bf16)f.x, (__bf16)f.y, (__bf16)f.z, (__bf16)f.w };
          *(v4bf*)(fr + 192 + 4 * i) = b;           // [192,320): raw x0
        }
      } else {
        const float4* xr = (const float4*)(x + (size_t)n * ROW + MUL0);
        #pragma unroll 2
        for (int g = 0; g < 16; ++g) {
          float4 p = xr[3*g+0], q = xr[3*g+1], r4 = xr[3*g+2];
          const float ax[4] = {p.x, p.w, q.z, r4.y};
          const float ay[4] = {p.y, q.x, q.w, r4.z};
          const float az[4] = {p.z, q.y, r4.x, r4.w};
          v4bf s, f0, f1, f2, c0, c1, c2;
          #pragma unroll
          for (int u = 0; u < 4; ++u) {
            s[u]  = (__bf16)(ax[u]*b0 + ay[u]*b1 + az[u]*b2);   // x1[u]·y1
            f0[u] = (__bf16)(y0 * ax[u]);
            f1[u] = (__bf16)(y0 * ay[u]);
            f2[u] = (__bf16)(y0 * az[u]);
            c0[u] = (__bf16)(ay[u]*b2 - az[u]*b1);              // cross(x1,y1)
            c1[u] = (__bf16)(az[u]*b0 - ax[u]*b2);
            c2[u] = (__bf16)(ax[u]*b1 - ay[u]*b0);
          }
          const int u0 = 4 * g;
          *(v4bf*)(fr + 128 + u0) = s;              // [128,192): s
          *(v4bf*)(fr + 320 + u0) = f0;             // [320+128k,+64): y0*x1_k
          *(v4bf*)(fr + 384 + u0) = c0;             // [384+128k,+64): cross_k
          *(v4bf*)(fr + 448 + u0) = f1;
          *(v4bf*)(fr + 512 + u0) = c1;
          *(v4bf*)(fr + 576 + u0) = f2;
          *(v4bf*)(fr + 640 + u0) = c2;
        }
      }
    }
    __syncthreads();

    // ---- Phases 2-4: WMMA GEMMs (EXEC all-ones: scalar predicate) ----
    if (active) {
      const int n0 = tile * 16;                      // scalar
      const bool fullTile = (n0 + 16) <= N;          // scalar compare
      const __bf16* frow = feat + m16 * FS;
      const int klo = hi * 8;

      // Single 64-bit base per lane; all stores use compile-time offsets.
      float* p0 = out + (size_t)(n0 + hi * 8) * ROW + m16;             // out0 base
      float* p1 = out + (size_t)(n0 + hi * 8) * ROW + MUL0 + 3 * m16;  // out1 base

      auto loadA = [&](int kb) -> v16bf {          // A 16x32 bf16 fragment
        v8bf lo = *(const v8bf*)(frow + kb + klo);
        v8bf hh = *(const v8bf*)(frow + kb + 16 + klo);
        return cat8(lo, hh);
      };
      auto loadB = [&](const __bf16* W, int stride, int kf, int nt) -> v16bf {
        const __bf16* p = W + (nt * 16 + m16) * stride + kf * 32 + hi * 8;
        return cat8(*(const v8bf*)p, *(const v8bf*)(p + 16));
      };

      // out0 (16x128): [x0*y0 ; s] (K=192) @ [W000 ; IS3*W110]
      v16bf A0[6];
      #pragma unroll
      for (int kf = 0; kf < 6; ++kf) A0[kf] = loadA(kf * 32);
      #pragma unroll
      for (int nt = 0; nt < 8; ++nt) {
        v8f acc = {0.f,0.f,0.f,0.f,0.f,0.f,0.f,0.f};
        #pragma unroll
        for (int kf = 0; kf < 6; ++kf)
          acc = wmma_bf16(A0[kf], loadB(WAt, WA_S, kf, nt), acc);
        const float bc = sBias[nt * 16 + m16];
        if (fullTile) {
          #pragma unroll
          for (int r = 0; r < 8; ++r)
            p0[r * ROW + nt * 16] = acc[r] + bc;    // immediate-offset stores
        } else {
          #pragma unroll
          for (int r = 0; r < 8; ++r)
            if (n0 + hi * 8 + r < N) p0[r * ROW + nt * 16] = acc[r] + bc;
        }
      }

      // t (16x64) = x0 @ W011 (K=128), kept in registers
      v16bf A1[4];
      #pragma unroll
      for (int kf = 0; kf < 4; ++kf) A1[kf] = loadA(192 + kf * 32);
      v8f tacc[4];
      #pragma unroll
      for (int nt = 0; nt < 4; ++nt) {
        v8f acc = {0.f,0.f,0.f,0.f,0.f,0.f,0.f,0.f};
        #pragma unroll
        for (int kf = 0; kf < 4; ++kf)
          acc = wmma_bf16(A1[kf], loadB(WBt, WB_S, kf, nt), acc);
        tacc[nt] = acc;
      }

      float y1r[8][3];
      #pragma unroll
      for (int r = 0; r < 8; ++r) {
        const int row = hi * 8 + r;
        y1r[r][0] = yW[row * 4 + 1];
        y1r[r][1] = yW[row * 4 + 2];
        y1r[r][2] = yW[row * 4 + 3];
      }

      // out1_k (16x64) = [y0*x1_k ; c_k] (K=128) @ [W101 ; IS2*W111] + t*y1k
      #pragma unroll
      for (int k = 0; k < 3; ++k) {
        v16bf A2[4];
        #pragma unroll
        for (int kf = 0; kf < 4; ++kf) A2[kf] = loadA(320 + 128 * k + kf * 32);
        #pragma unroll
        for (int nt = 0; nt < 4; ++nt) {
          v8f acc = {0.f,0.f,0.f,0.f,0.f,0.f,0.f,0.f};
          #pragma unroll
          for (int kf = 0; kf < 4; ++kf)
            acc = wmma_bf16(A2[kf], loadB(WCt, WB_S, kf, nt), acc);
          if (fullTile) {
            #pragma unroll
            for (int r = 0; r < 8; ++r)
              p1[r * ROW + 48 * nt + k] = acc[r] + tacc[nt][r] * y1r[r][k];
          } else {
            #pragma unroll
            for (int r = 0; r < 8; ++r)
              if (n0 + hi * 8 + r < N)
                p1[r * ROW + 48 * nt + k] = acc[r] + tacc[nt][r] * y1r[r][k];
          }
        }
      }
    }
    __syncthreads();
  }
}

extern "C" void kernel_launch(void* const* d_in, const int* in_sizes, int n_in,
                              void* d_out, int out_size, void* d_ws, size_t ws_size,
                              hipStream_t stream) {
  const float* x    = (const float*)d_in[0];
  const float* y    = (const float*)d_in[1];
  const float* w000 = (const float*)d_in[2];
  const float* w011 = (const float*)d_in[3];
  const float* w101 = (const float*)d_in[4];
  const float* w110 = (const float*)d_in[5];
  const float* w111 = (const float*)d_in[6];
  const float* bias = (const float*)d_in[7];
  float* out = (float*)d_out;

  const int N      = in_sizes[0] / ROW;
  const int ntiles = (N + 15) / 16;
  const int blocks = 256;
  const int totalWaves = blocks * WAVES;
  const int iters  = (ntiles + totalWaves - 1) / totalWaves;

  const size_t smem = (size_t)(128 * WA_S + 2 * 64 * WB_S) * sizeof(__bf16)
                    + 128 * sizeof(float)
                    + (size_t)WAVES * 64 * sizeof(float)
                    + (size_t)WAVES * 16 * FS * sizeof(__bf16);

  tpr_kernel<<<blocks, 256, smem, stream>>>(x, y, w000, w011, w101, w110, w111,
                                            bias, out, N, ntiles, iters);
}
